// bindingPlaceDetector_19688130084894
// MI455X (gfx1250) — compile-verified, matrix operand order checked by reading, not run on hardware
//
#include <hip/hip_runtime.h>
#include <hip/hip_bf16.h>

// ---------------------------------------------------------------------------
// bindingPlaceDetector pipeline for MI455X (gfx1250, wave32, WMMA)
//
//   Heavy path: WLC masked matmuls done with v_wmma_f32_16x16x32_bf16.
//   A-tiles (edge-type masks) generated on the fly from ligand_structure s;
//   one f32 accumulator per edge type k; per-feature K scale folded at end.
// ---------------------------------------------------------------------------

typedef __attribute__((ext_vector_type(16))) __bf16 v16bf;   // 8 VGPRs
typedef __attribute__((ext_vector_type(8)))  __bf16 bf16x8;  // 16 bytes
typedef __attribute__((ext_vector_type(8)))  float  v8f;     // 8 VGPRs

#define B_  16
#define NN  1024
#define FQ  35
#define FP  48           // padded feature width (3 WMMA n-tiles)
#define NF  (FP * NN)    // elements per channel image (feature-major)
#define HID 2000
#define LIN 1610

__device__ __forceinline__ float leakyf(float v) { return v >= 0.f ? v : 0.01f * v; }

// ---------------------------------------------------------------------------
// Encode labelsList [B,N,35] f32 -> feature-major bf16 [B][48][1024], zero pad
// ---------------------------------------------------------------------------
__global__ void encode_x_kernel(const float* __restrict__ L, __bf16* __restrict__ X) {
    size_t i = (size_t)blockIdx.x * blockDim.x + threadIdx.x;
    const size_t total = (size_t)B_ * FP * NN;
    if (i >= total) return;
    int n = (int)(i % NN);
    size_t r = i / NN;
    int f = (int)(r % FP);
    int b = (int)(r / FP);
    float v = (f < FQ) ? L[((size_t)b * NN + n) * FQ + f] : 0.f;
    X[i] = (__bf16)v;
}

// ---------------------------------------------------------------------------
// One WLC step over all samples and channel-combos:
//   xout[z] = maybe_leaky( xin[z>>shift] + sum_k (A_k @ xin) * K[kc,t,k,:] )
// grid = (NN/128, B, combos); block = 256 (8 waves, 16 rows of M per wave)
// ---------------------------------------------------------------------------
__global__ __launch_bounds__(256)
void wlc_step_kernel(const float* __restrict__ S,          // [B][N][N]
                     const __bf16* __restrict__ Xin, int in_nch, int in_shift,
                     __bf16* __restrict__ Xout,            // [B][gridDim.z][FP][NN]
                     const float* __restrict__ Kbase,      // [2][2][4][35]
                     int t, int stageB, int apply_leaky) {
    const int wave   = threadIdx.x >> 5;
    const int lane   = threadIdx.x & 31;
    const int laneHi = lane >> 4;
    const int lane16 = lane & 15;
    const int b      = blockIdx.y;
    const int z      = blockIdx.z;
    const int outNch = gridDim.z;
    const int m0     = blockIdx.x * 128 + wave * 16;

    const int in_ch = z >> in_shift;
    const __bf16* xin  = Xin  + ((size_t)b * in_nch + in_ch) * (size_t)NF;
    __bf16*       xout = Xout + ((size_t)b * outNch + z)     * (size_t)NF;
    const int kc = stageB ? (z & 1) : z;
    const float* Kp = Kbase + (size_t)(kc * 2 + t) * 4 * FQ;

    // A fragment row for this lane (lanes 0-15 and 16-31 cover the same 16 rows)
    const float* sA = S + ((size_t)b * NN + (m0 + lane16)) * NN;

    v8f acc[3][4];
#pragma unroll
    for (int nt = 0; nt < 3; ++nt)
#pragma unroll
        for (int k = 0; k < 4; ++k)
#pragma unroll
            for (int r = 0; r < 8; ++r) acc[nt][k][r] = 0.f;

    for (int kc0 = 0; kc0 < NN; kc0 += 32) {
        const int jb = kc0 + (laneHi << 3);   // K-dim base for this lane half
        // --- load 16 s values for this lane's A row (K = jb..jb+7, jb+16..jb+23)
        float sv[16];
        {
            const float4 a0 = *(const float4*)(sA + jb);
            const float4 a1 = *(const float4*)(sA + jb + 4);
            const float4 a2 = *(const float4*)(sA + jb + 16);
            const float4 a3 = *(const float4*)(sA + jb + 20);
            sv[0] = a0.x;  sv[1] = a0.y;  sv[2] = a0.z;  sv[3] = a0.w;
            sv[4] = a1.x;  sv[5] = a1.y;  sv[6] = a1.z;  sv[7] = a1.w;
            sv[8] = a2.x;  sv[9] = a2.y;  sv[10] = a2.z; sv[11] = a2.w;
            sv[12] = a3.x; sv[13] = a3.y; sv[14] = a3.z; sv[15] = a3.w;
        }
        // --- build the 4 edge-type mask fragments (exact 0/1 in bf16)
        v16bf afr[4];
#pragma unroll
        for (int k = 0; k < 4; ++k) {
            const float tv = (float)(k + 1);
#pragma unroll
            for (int e = 0; e < 16; ++e)
                afr[k][e] = (sv[e] == tv) ? (__bf16)1.0f : (__bf16)0.0f;
        }
        // --- B fragments (feature-major storage -> contiguous K loads) + WMMA
#pragma unroll
        for (int nt = 0; nt < 3; ++nt) {
            const __bf16* xc = xin + (size_t)(nt * 16 + lane16) * NN;
            bf16x8 blo = *(const bf16x8*)(xc + jb);
            bf16x8 bhi = *(const bf16x8*)(xc + jb + 16);
            v16bf bfr;
#pragma unroll
            for (int e = 0; e < 8; ++e) { bfr[e] = blo[e]; bfr[e + 8] = bhi[e]; }
#pragma unroll
            for (int k = 0; k < 4; ++k)
                acc[nt][k] = __builtin_amdgcn_wmma_f32_16x16x32_bf16(
                    false, afr[k], false, bfr, (short)0, acc[nt][k], false, false);
        }
    }

    // --- fold per-feature K weights, add residual, activation, write transposed
#pragma unroll
    for (int nt = 0; nt < 3; ++nt) {
        const int f = nt * 16 + lane16;
        float kf[4];
#pragma unroll
        for (int k = 0; k < 4; ++k) kf[k] = (f < FQ) ? Kp[k * FQ + f] : 0.f;
        const size_t base = (size_t)f * NN + m0 + laneHi * 8;   // D: M = r + 8*laneHi
        bf16x8 xold = *(const bf16x8*)(xin + base);
        bf16x8 outv;
#pragma unroll
        for (int r = 0; r < 8; ++r) {
            float v = (float)xold[r];
            v += acc[nt][0][r] * kf[0] + acc[nt][1][r] * kf[1] +
                 acc[nt][2][r] * kf[2] + acc[nt][3][r] * kf[3];
            if (apply_leaky) v = leakyf(v);
            outv[r] = (__bf16)v;
        }
        *(bf16x8*)(xout + base) = outv;
    }
}

// ---------------------------------------------------------------------------
// SPP max over c2 [B][4][FP][NN] (image is [C=4, H=1024(n), W=35(f)]),
// levels 1..10; one wave per output element; writes h[b][ch*385 + o]
// ---------------------------------------------------------------------------
__global__ __launch_bounds__(256)
void spp_max_kernel(const __bf16* __restrict__ C2, float* __restrict__ H) {
    const int wgid = blockIdx.x * 8 + (threadIdx.x >> 5);
    const int lane = threadIdx.x & 31;
    const int total = B_ * 4 * 385;
    if (wgid >= total) return;
    const int b    = wgid / (4 * 385);
    const int rem  = wgid % (4 * 385);
    const int ch   = rem / 385;
    const int orig = rem % 385;

    int o = orig, p = 1;
    while (o >= p * p) { o -= p * p; ++p; }

    int kh = (NN + p - 1) / p, kw = (FQ + p - 1) / p;
    int ph = kh * p - NN, pw = kw * p - FQ;
    if (ph * 2 > kh) ph = kh / 2;
    if (pw * 2 > kw) pw = kw / 2;
    const int out_h = (NN + 2 * ph - kh) / kh + 1;
    const int out_w = (FQ + 2 * pw - kw) / kw + 1;

    float res;
    if (o >= out_h * out_w) {
        res = 0.f;   // jnp.pad zero-fill of the flattened level
    } else {
        const int oy = o / out_w, ox = o % out_w;
        const int y0 = oy * kh - ph, x0 = ox * kw - pw;
        const __bf16* img = C2 + (size_t)(b * 4 + ch) * (size_t)NF;
        float m = -__builtin_inff();
        const int win = kh * kw;
        for (int e = lane; e < win; e += 32) {
            const int y = y0 + e / kw, x = x0 + e % kw;
            if (y >= 0 && y < NN && x >= 0 && x < FQ)
                m = fmaxf(m, (float)img[(size_t)x * NN + y]);
        }
        for (int off = 16; off > 0; off >>= 1)
            m = fmaxf(m, __shfl_xor(m, off, 32));
        res = m;
    }
    if (lane == 0) H[(size_t)b * LIN + ch * 385 + orig] = res;
}

// ---------------------------------------------------------------------------
// Fused conv3x3(1->2)+leaky -> conv3x3(2->5)+leaky -> avg-pool partial sums
// grid = (64, 64, B); block = (16,16). gsum[b][5][14] accumulated atomically.
// ---------------------------------------------------------------------------
__global__ __launch_bounds__(256)
void conv_spp_kernel(const float* __restrict__ S,
                     const float* __restrict__ w1, const float* __restrict__ bb1,
                     const float* __restrict__ w2, const float* __restrict__ bb2,
                     float* __restrict__ gsum) {
    __shared__ float st[20][20];
    __shared__ float c1t[2][18][18];
    __shared__ float ws1[18], wb1[2], ws2[90], wb2[5];
    __shared__ float cell[5][14];

    const int b  = blockIdx.z;
    const int Y0 = blockIdx.y * 16, X0 = blockIdx.x * 16;
    const int tid = threadIdx.y * 16 + threadIdx.x;

    if (tid < 18) ws1[tid] = w1[tid];
    else if (tid < 20) wb1[tid - 18] = bb1[tid - 18];
    if (tid >= 32 && tid < 122) ws2[tid - 32] = w2[tid - 32];
    if (tid >= 128 && tid < 133) wb2[tid - 128] = bb2[tid - 128];
    if (tid >= 160 && tid < 230) ((float*)cell)[tid - 160] = 0.f;

    for (int i = tid; i < 400; i += 256) {
        const int ry = i / 20, rx = i % 20;
        const int y = Y0 + ry, x = X0 + rx;
        st[ry][rx] = (y < NN && x < NN) ? S[((size_t)b * NN + y) * NN + x] : 0.f;
    }
    __syncthreads();

    for (int i = tid; i < 2 * 18 * 18; i += 256) {
        const int c = i / 324, r = i % 324, ry = r / 18, rx = r % 18;
        float v = wb1[c];
#pragma unroll
        for (int di = 0; di < 3; ++di)
#pragma unroll
            for (int dj = 0; dj < 3; ++dj)
                v += st[ry + di][rx + dj] * ws1[c * 9 + di * 3 + dj];
        c1t[c][ry][rx] = leakyf(v);
    }
    __syncthreads();

    const int y = Y0 + threadIdx.y, x = X0 + threadIdx.x;
    if (y < 1020 && x < 1020) {
        const int cell1 = 0;
        const int cell2 = 1 + (y / 510) * 2 + (x / 510);
        const int cell3 = 5 + (y / 340) * 3 + (x / 340);
#pragma unroll
        for (int oc = 0; oc < 5; ++oc) {
            float v = wb2[oc];
#pragma unroll
            for (int ic = 0; ic < 2; ++ic)
#pragma unroll
                for (int di = 0; di < 3; ++di)
#pragma unroll
                    for (int dj = 0; dj < 3; ++dj)
                        v += c1t[ic][threadIdx.y + di][threadIdx.x + dj] *
                             ws2[((oc * 2 + ic) * 3 + di) * 3 + dj];
            v = leakyf(v);
            atomicAdd(&cell[oc][cell1], v);
            atomicAdd(&cell[oc][cell2], v);
            atomicAdd(&cell[oc][cell3], v);
        }
    }
    __syncthreads();
    if (tid < 70) atomicAdd(&gsum[(size_t)b * 70 + tid], ((float*)cell)[tid]);
}

__global__ void zero_kernel(float* __restrict__ p, int n) {
    const int i = blockIdx.x * blockDim.x + threadIdx.x;
    if (i < n) p[i] = 0.f;
}

__global__ void stru_finalize_kernel(const float* __restrict__ gsum, float* __restrict__ H) {
    const int i = blockIdx.x * blockDim.x + threadIdx.x;
    if (i >= B_ * 70) return;
    const int b = i / 70, r = i % 70;
    const int ch = r / 14, idx = r % 14;
    const int p = (idx == 0) ? 1 : (idx < 5 ? 2 : 3);
    const float k = (float)(1020 / p);
    H[(size_t)b * LIN + 1540 + ch * 14 + idx] = gsum[i] / (k * k);
}

// ---------------------------------------------------------------------------
// FC layer: one wave per (b, o); Y[b][o] = act( X[b,:] . W[o,:] + bias[o] )
// ---------------------------------------------------------------------------
__global__ __launch_bounds__(256)
void fc_kernel(const float* __restrict__ X, const float* __restrict__ W,
               const float* __restrict__ bias, float* __restrict__ Y,
               int INn, int OUT, int applyLeaky) {
    const int w    = (int)(((size_t)blockIdx.x * blockDim.x + threadIdx.x) >> 5);
    const int lane = threadIdx.x & 31;
    if (w >= B_ * OUT) return;
    const int b = w / OUT, o = w % OUT;
    const float* xr = X + (size_t)b * INn;
    const float* wr = W + (size_t)o * INn;
    float sum = 0.f;
    for (int i = lane; i < INn; i += 32) sum += xr[i] * wr[i];
    for (int off = 16; off > 0; off >>= 1) sum += __shfl_xor(sum, off, 32);
    if (lane == 0) {
        float v = sum + bias[o];
        if (applyLeaky) v = leakyf(v);
        Y[(size_t)b * OUT + o] = v;
    }
}

// ---------------------------------------------------------------------------
extern "C" void kernel_launch(void* const* d_in, const int* in_sizes, int n_in,
                              void* d_out, int out_size, void* d_ws, size_t ws_size,
                              hipStream_t stream) {
    (void)in_sizes; (void)n_in; (void)out_size; (void)ws_size;
    const float* L   = (const float*)d_in[0];
    const float* S   = (const float*)d_in[1];
    const float* K1  = (const float*)d_in[2];
    const float* K2  = (const float*)d_in[3];
    const float* cw1 = (const float*)d_in[4];
    const float* cb1 = (const float*)d_in[5];
    const float* cw2 = (const float*)d_in[6];
    const float* cb2 = (const float*)d_in[7];
    const float* W1  = (const float*)d_in[8];   const float* b1 = (const float*)d_in[9];
    const float* W2  = (const float*)d_in[10];  const float* b2 = (const float*)d_in[11];
    const float* W3  = (const float*)d_in[12];  const float* b3 = (const float*)d_in[13];
    const float* W4  = (const float*)d_in[14];  const float* b4 = (const float*)d_in[15];
    const float* W5  = (const float*)d_in[16];  const float* b5 = (const float*)d_in[17];
    float* out = (float*)d_out;

    // ---- workspace bump allocator (256B aligned)
    char* ws = (char*)d_ws;
    size_t off = 0;
    auto alloc = [&](size_t bytes) -> void* {
        off = (off + 255) & ~(size_t)255;
        void* p = ws + off;
        off += bytes;
        return p;
    };
    __bf16* xenc = (__bf16*)alloc((size_t)B_ * 1 * NF * sizeof(__bf16));
    __bf16* bufA = (__bf16*)alloc((size_t)B_ * 2 * NF * sizeof(__bf16));
    __bf16* c1   = (__bf16*)alloc((size_t)B_ * 2 * NF * sizeof(__bf16));
    __bf16* bufB = (__bf16*)alloc((size_t)B_ * 4 * NF * sizeof(__bf16));
    __bf16* c2   = (__bf16*)alloc((size_t)B_ * 4 * NF * sizeof(__bf16));
    float*  h0   = (float*)alloc((size_t)B_ * LIN * sizeof(float));
    float*  hb1  = (float*)alloc((size_t)B_ * HID * sizeof(float));
    float*  hb2  = (float*)alloc((size_t)B_ * HID * sizeof(float));
    float*  gsum = (float*)alloc((size_t)B_ * 70 * sizeof(float));

    // ---- 0) zero the pooling accumulators
    zero_kernel<<<dim3((B_ * 70 + 255) / 256), dim3(256), 0, stream>>>(gsum, B_ * 70);

    // ---- 1) encode features to bf16, feature-major, padded to 48
    {
        const size_t total = (size_t)B_ * FP * NN;
        encode_x_kernel<<<dim3((unsigned)((total + 255) / 256)), dim3(256), 0, stream>>>(L, xenc);
    }

    // ---- 2) WLC stage A (K1): 2 channels, 2 scan steps
    wlc_step_kernel<<<dim3(8, B_, 2), dim3(256), 0, stream>>>(
        S, xenc, 1, 2, bufA, K1, /*t=*/0, /*stageB=*/0, /*leaky=*/0);
    wlc_step_kernel<<<dim3(8, B_, 2), dim3(256), 0, stream>>>(
        S, bufA, 2, 0, c1, K1, /*t=*/1, /*stageB=*/0, /*leaky=*/1);

    // ---- 3) WLC stage B (K2): 4 combos (c1 channel = z>>1, K channel = z&1)
    wlc_step_kernel<<<dim3(8, B_, 4), dim3(256), 0, stream>>>(
        S, c1, 2, 1, bufB, K2, /*t=*/0, /*stageB=*/1, /*leaky=*/0);
    wlc_step_kernel<<<dim3(8, B_, 4), dim3(256), 0, stream>>>(
        S, bufB, 4, 0, c2, K2, /*t=*/1, /*stageB=*/1, /*leaky=*/1);

    // ---- 4) SPP max over c2 -> h0[:, 0:1540]
    {
        const int waves = B_ * 4 * 385;
        spp_max_kernel<<<dim3((waves + 7) / 8), dim3(256), 0, stream>>>(c2, h0);
    }

    // ---- 5) structure branch: fused convs + avg-pool sums, then finalize
    conv_spp_kernel<<<dim3(64, 64, B_), dim3(16, 16), 0, stream>>>(
        S, cw1, cb1, cw2, cb2, gsum);
    stru_finalize_kernel<<<dim3((B_ * 70 + 255) / 256), dim3(256), 0, stream>>>(gsum, h0);

    // ---- 6) MLP head
    auto fc = [&](const float* X, const float* W, const float* bias, float* Y,
                  int INn, int OUT, int lk) {
        const int waves = B_ * OUT;
        fc_kernel<<<dim3((waves + 7) / 8), dim3(256), 0, stream>>>(X, W, bias, Y, INn, OUT, lk);
    };
    fc(h0,  W1, b1, hb1, LIN, HID, 1);
    fc(hb1, W2, b2, hb2, HID, HID, 1);
    fc(hb2, W3, b3, hb1, HID, HID, 1);
    fc(hb1, W4, b4, hb2, HID, HID, 1);
    fc(hb2, W5, b5, out, HID, 2,   0);
}
